// MikesGraphNet_25099788877915
// MI455X (gfx1250) — matrix operand order repaired and checked by reading, not compiled, not run on hardware
//
#include <hip/hip_runtime.h>
#include <hip/hip_bf16.h>
#include <math.h>

// ---------------------------------------------------------------------------
// MikesGraphNet forward for MI455X (gfx1250, wave32, WMMA).
// All dense GEMMs run on v_wmma_f32_16x16x32_bf16 (bf16 A/B, f32 accum).
// Edge MLP fully fused (rbf->edge_attr, gathers, mp1+LN+gelu, mp2, scatter)
// so no E x {64,192} intermediates ever touch HBM.
// ---------------------------------------------------------------------------

#define NN    100000
#define EE    400000
#define HH    128
#define FF    64
#define RR    12
#define LL    2
#define NFEAT 9
#define EMBD  32
#define KEMB  (EMBD + NFEAT)   // 41
#define VOCAB 100

typedef __attribute__((ext_vector_type(16))) __bf16 v16bf;
typedef __attribute__((ext_vector_type(8)))  float  v8f;

__device__ __forceinline__ float gelu_f(float x) {
  return 0.5f * x * (1.0f + erff(x * 0.70710678118654752f));
}

// A-fragment: 16x32 bf16 tile from LDS (row stride sA elems), K-chunk base kb.
// Contiguous k order (matches frag_bt): lane m<16 holds row M=m; element e -> K = hi*16 + e.
__device__ __forceinline__ v16bf frag_a(const __bf16* A, int sA, int kb, int lane) {
  int m = lane & 15, hi = lane >> 4;
  v16bf a;
#pragma unroll
  for (int e = 0; e < 16; ++e) {
    int k = kb + hi * 16 + e;
    a[e] = A[m * sA + k];
  }
  return a;
}

// B-fragment from transposed weights BT[n][k] (row stride sB elems).
// Contiguous k order (matches frag_a): lane holds col N=lane&15; element e -> K = kb + hi*16 + e.
__device__ __forceinline__ v16bf frag_bt(const __bf16* BT, int sB, int nb, int kb, int lane) {
  int n = lane & 15, hi = lane >> 4;
  int base = (nb + n) * sB + kb + hi * 16;
  v16bf b;
#pragma unroll
  for (int e = 0; e < 16; ++e) b[e] = BT[base + e];
  return b;
}

#define WMMA_BF16(a, b, c) \
  __builtin_amdgcn_wmma_f32_16x16x32_bf16(false, (a), false, (b), (short)0, (c), false, false)

// Sum across the 16 lanes of a half-wave (masks 1..8 stay inside the group).
__device__ __forceinline__ float half16_sum(float v) {
#pragma unroll
  for (int m = 8; m >= 1; m >>= 1) v += __shfl_xor(v, m, 32);
  return v;
}

#define LDS_FENCE() asm volatile("s_wait_dscnt 0" ::: "memory")

// ---------------------------------------------------------------------------
// Bessel radial basis with polynomial envelope (P=6 => -28 d^5 + 48 d^6 - 21 d^7)
// ---------------------------------------------------------------------------
__global__ void k_rbf(const int* __restrict__ eidx, const float* __restrict__ pos,
                      const float* __restrict__ freq, float* __restrict__ rbf) {
  int e = blockIdx.x * blockDim.x + threadIdx.x;
  if (e >= EE) return;
  int i = eidx[e], j = eidx[EE + e];
  float dx = pos[3 * i]     - pos[3 * j];
  float dy = pos[3 * i + 1] - pos[3 * j + 1];
  float dz = pos[3 * i + 2] - pos[3 * j + 2];
  float dist = sqrtf(dx * dx + dy * dy + dz * dz + 1e-8f);
  float d = dist * (1.0f / 5.0f);
  float d2 = d * d;
  float d5 = d2 * d2 * d;
  float env = 1.0f / d + d5 * (-28.0f + d * (48.0f + d * (-21.0f)));
#pragma unroll
  for (int r = 0; r < RR; ++r)
    rbf[(size_t)e * RR + r] = env * sinf(freq[r] * d);
}

// ---------------------------------------------------------------------------
// Embedding: x = gelu([emb_table[type], feats] @ emb_W + b)   (K=41, VALU)
// ---------------------------------------------------------------------------
__global__ void k_embed(const int* __restrict__ types, const float* __restrict__ feats,
                        const float* __restrict__ table, const float* __restrict__ W,
                        const float* __restrict__ b, float* __restrict__ x) {
  __shared__ float sW[KEMB * HH];
  __shared__ float sT[VOCAB * EMBD];
  __shared__ float sB[HH];
  for (int idx = threadIdx.x; idx < KEMB * HH; idx += blockDim.x) sW[idx] = W[idx];
  for (int idx = threadIdx.x; idx < VOCAB * EMBD; idx += blockDim.x) sT[idx] = table[idx];
  for (int idx = threadIdx.x; idx < HH; idx += blockDim.x) sB[idx] = b[idx];
  __syncthreads();
  int t = threadIdx.x;  // 0..127 -> output column
  for (int row = blockIdx.x; row < NN; row += gridDim.x) {
    int ty = types[row];
    const float* emb = &sT[ty * EMBD];
    float acc = sB[t];
#pragma unroll
    for (int k = 0; k < EMBD; ++k) acc += emb[k] * sW[k * HH + t];
#pragma unroll
    for (int k = 0; k < NFEAT; ++k) acc += feats[(size_t)row * NFEAT + k] * sW[(EMBD + k) * HH + t];
    x[(size_t)row * HH + t] = gelu_f(acc);
  }
}

__global__ void k_zero(float* __restrict__ p, int n) {
  int i = blockIdx.x * blockDim.x + threadIdx.x;
  if (i < n) p[i] = 0.0f;
}

// ---------------------------------------------------------------------------
// m_in = LN(x @ n2m_W + b)      [N,128] -> [N,64], WMMA + in-register LN
// ---------------------------------------------------------------------------
#define SA_N2M 136
__global__ void k_n2m(const float* __restrict__ x, const float* __restrict__ W,
                      const float* __restrict__ bias, const float* __restrict__ g,
                      const float* __restrict__ be, float* __restrict__ m_in) {
  __shared__ __bf16 sBT[FF * HH];                 // W^T : [n=64][k=128]
  __shared__ float sBias[FF], sG[FF], sBe[FF];
  __shared__ __bf16 sA[4][16 * SA_N2M];
  for (int idx = threadIdx.x; idx < FF * HH; idx += blockDim.x) {
    int k = idx / FF, n = idx % FF;
    sBT[n * HH + k] = (__bf16)W[idx];
  }
  for (int idx = threadIdx.x; idx < FF; idx += blockDim.x) {
    sBias[idx] = bias[idx]; sG[idx] = g[idx]; sBe[idx] = be[idx];
  }
  __syncthreads();
  int lane = threadIdx.x & 31, wave = threadIdx.x >> 5, nw = blockDim.x >> 5;
  __bf16* At = sA[wave];
  for (int tile = blockIdx.x * nw + wave; tile < NN / 16; tile += gridDim.x * nw) {
    const float* xrow = x + (size_t)tile * 16 * HH;
    for (int idx = lane; idx < 16 * HH; idx += 32) {
      int r = idx >> 7, c = idx & 127;
      At[r * SA_N2M + c] = (__bf16)xrow[r * HH + c];
    }
    LDS_FENCE();
    v8f acc[4] = {};
#pragma unroll
    for (int kc = 0; kc < 4; ++kc) {
      v16bf a = frag_a(At, SA_N2M, kc * 32, lane);
#pragma unroll
      for (int nt = 0; nt < 4; ++nt)
        acc[nt] = WMMA_BF16(a, frag_bt(sBT, HH, nt * 16, kc * 32, lane), acc[nt]);
    }
    int n = lane & 15, hi = lane >> 4;
#pragma unroll
    for (int r = 0; r < 8; ++r) {
      float v[4], s = 0.f, s2 = 0.f;
#pragma unroll
      for (int nt = 0; nt < 4; ++nt) {
        v[nt] = acc[nt][r] + sBias[nt * 16 + n];
        s += v[nt]; s2 += v[nt] * v[nt];
      }
      s = half16_sum(s); s2 = half16_sum(s2);
      float mu = s * (1.0f / FF);
      float rs = rsqrtf(s2 * (1.0f / FF) - mu * mu + 1e-5f);
      int row = tile * 16 + r + 8 * hi;
#pragma unroll
      for (int nt = 0; nt < 4; ++nt) {
        int c = nt * 16 + n;
        m_in[(size_t)row * FF + c] = (v[nt] - mu) * rs * sG[c] + sBe[c];
      }
    }
    LDS_FENCE();
  }
}

// ---------------------------------------------------------------------------
// Fused edge MLP: edge_attr(rbf) + gather(m_in[i],m_in[j]) -> mp1 -> LN -> gelu
//                 -> mp2 -> atomic scatter-add into agg[j]
// ---------------------------------------------------------------------------
#define SA_E1 200
#define SA_E2 72
__global__ void k_edge(const int* __restrict__ eidx, const float* __restrict__ rbf,
                       const float* __restrict__ m_in,
                       const float* __restrict__ r2mW, const float* __restrict__ r2mb,
                       const float* __restrict__ mp1W, const float* __restrict__ mp1b,
                       const float* __restrict__ lng, const float* __restrict__ lnb,
                       const float* __restrict__ mp2W, const float* __restrict__ mp2b,
                       float* __restrict__ agg) {
  __shared__ __bf16 sBT1[FF * 192];               // mp1_W^T : [n=64][k=192]
  __shared__ __bf16 sBT2[FF * FF];                // mp2_W^T : [n=64][k=64]
  __shared__ float  sR2M[RR * FF];
  __shared__ float  sR2Mb[FF], sMP1b[FF], sLNg[FF], sLNb[FF], sMP2b[FF];
  __shared__ __bf16 sCat[2][16 * SA_E1];
  __shared__ __bf16 sGm[2][16 * SA_E2];
  __shared__ int    sI[2][16], sJ[2][16];
  for (int idx = threadIdx.x; idx < 192 * FF; idx += blockDim.x) {
    int k = idx / FF, n = idx % FF;
    sBT1[n * 192 + k] = (__bf16)mp1W[idx];
  }
  for (int idx = threadIdx.x; idx < FF * FF; idx += blockDim.x) {
    int k = idx / FF, n = idx % FF;
    sBT2[n * FF + k] = (__bf16)mp2W[idx];
  }
  for (int idx = threadIdx.x; idx < RR * FF; idx += blockDim.x) sR2M[idx] = r2mW[idx];
  for (int idx = threadIdx.x; idx < FF; idx += blockDim.x) {
    sR2Mb[idx] = r2mb[idx]; sMP1b[idx] = mp1b[idx];
    sLNg[idx] = lng[idx];   sLNb[idx] = lnb[idx]; sMP2b[idx] = mp2b[idx];
  }
  __syncthreads();
  int lane = threadIdx.x & 31, wave = threadIdx.x >> 5, nw = blockDim.x >> 5;
  __bf16* cat = sCat[wave];
  __bf16* gt  = sGm[wave];
  int *iI = sI[wave], *iJ = sJ[wave];
  for (int tile = blockIdx.x * nw + wave; tile < EE / 16; tile += gridDim.x * nw) {
    int e0 = tile * 16;
    if (lane < 16) { iI[lane] = eidx[e0 + lane]; iJ[lane] = eidx[EE + e0 + lane]; }
    LDS_FENCE();
    // gather m_in rows into cat columns [0,64) and [64,128)
    for (int idx = lane; idx < 16 * FF; idx += 32) {
      int r = idx >> 6, c = idx & 63;
      cat[r * SA_E1 + c]      = (__bf16)m_in[(size_t)iI[r] * FF + c];
      cat[r * SA_E1 + 64 + c] = (__bf16)m_in[(size_t)iJ[r] * FF + c];
    }
    // edge_attr = rbf @ r2m_W + b into cat columns [128,192)
    for (int idx = lane; idx < 16 * FF; idx += 32) {
      int r = idx >> 6, c = idx & 63;
      const float* rb = rbf + (size_t)(e0 + r) * RR;
      float a = sR2Mb[c];
#pragma unroll
      for (int k = 0; k < RR; ++k) a += rb[k] * sR2M[k * FF + c];
      cat[r * SA_E1 + 128 + c] = (__bf16)a;
    }
    LDS_FENCE();
    // GEMM1: [16x192] @ [192x64]
    v8f acc[4] = {};
#pragma unroll
    for (int kc = 0; kc < 6; ++kc) {
      v16bf a = frag_a(cat, SA_E1, kc * 32, lane);
#pragma unroll
      for (int nt = 0; nt < 4; ++nt)
        acc[nt] = WMMA_BF16(a, frag_bt(sBT1, 192, nt * 16, kc * 32, lane), acc[nt]);
    }
    int n = lane & 15, hi = lane >> 4;
#pragma unroll
    for (int r = 0; r < 8; ++r) {
      float v[4], s = 0.f, s2 = 0.f;
#pragma unroll
      for (int nt = 0; nt < 4; ++nt) {
        v[nt] = acc[nt][r] + sMP1b[nt * 16 + n];
        s += v[nt]; s2 += v[nt] * v[nt];
      }
      s = half16_sum(s); s2 = half16_sum(s2);
      float mu = s * (1.0f / FF);
      float rs = rsqrtf(s2 * (1.0f / FF) - mu * mu + 1e-5f);
      int row = r + 8 * hi;
#pragma unroll
      for (int nt = 0; nt < 4; ++nt) {
        int c = nt * 16 + n;
        gt[row * SA_E2 + c] = (__bf16)gelu_f((v[nt] - mu) * rs * sLNg[c] + sLNb[c]);
      }
    }
    LDS_FENCE();
    // GEMM2: [16x64] @ [64x64], then scatter-add by j
    v8f acc2[4] = {};
#pragma unroll
    for (int kc = 0; kc < 2; ++kc) {
      v16bf a = frag_a(gt, SA_E2, kc * 32, lane);
#pragma unroll
      for (int nt = 0; nt < 4; ++nt)
        acc2[nt] = WMMA_BF16(a, frag_bt(sBT2, FF, nt * 16, kc * 32, lane), acc2[nt]);
    }
#pragma unroll
    for (int r = 0; r < 8; ++r) {
      int row = r + 8 * hi;
      int jn = iJ[row];
#pragma unroll
      for (int nt = 0; nt < 4; ++nt) {
        int c = nt * 16 + n;
        unsafeAtomicAdd(&agg[(size_t)jn * FF + c], acc2[nt][r] + sMP2b[c]);
      }
    }
    LDS_FENCE();
  }
}

// ---------------------------------------------------------------------------
// x += agg @ m2n_W     [N,64] -> [N,128]
// ---------------------------------------------------------------------------
#define SA_M2N 72
__global__ void k_m2n(const float* __restrict__ agg, const float* __restrict__ W,
                      float* __restrict__ x) {
  __shared__ __bf16 sBT[HH * FF];                 // W^T : [n=128][k=64]
  __shared__ __bf16 sA[4][16 * SA_M2N];
  for (int idx = threadIdx.x; idx < FF * HH; idx += blockDim.x) {
    int k = idx / HH, n = idx % HH;
    sBT[n * FF + k] = (__bf16)W[idx];
  }
  __syncthreads();
  int lane = threadIdx.x & 31, wave = threadIdx.x >> 5, nw = blockDim.x >> 5;
  __bf16* At = sA[wave];
  for (int tile = blockIdx.x * nw + wave; tile < NN / 16; tile += gridDim.x * nw) {
    const float* arow = agg + (size_t)tile * 16 * FF;
    for (int idx = lane; idx < 16 * FF; idx += 32) {
      int r = idx >> 6, c = idx & 63;
      At[r * SA_M2N + c] = (__bf16)arow[r * FF + c];
    }
    LDS_FENCE();
    v8f acc[8] = {};
#pragma unroll
    for (int kc = 0; kc < 2; ++kc) {
      v16bf a = frag_a(At, SA_M2N, kc * 32, lane);
#pragma unroll
      for (int nt = 0; nt < 8; ++nt)
        acc[nt] = WMMA_BF16(a, frag_bt(sBT, FF, nt * 16, kc * 32, lane), acc[nt]);
    }
    int n = lane & 15, hi = lane >> 4;
#pragma unroll
    for (int r = 0; r < 8; ++r) {
      int row = tile * 16 + r + 8 * hi;
#pragma unroll
      for (int nt = 0; nt < 8; ++nt) {
        int c = nt * 16 + n;
        x[(size_t)row * HH + c] += acc[nt][r];
      }
    }
    LDS_FENCE();
  }
}

// ---------------------------------------------------------------------------
// h = gelu(LN(x) @ fc1_W + b)   (bf16 intermediate),  then  x += h @ fc2_W + b
// ---------------------------------------------------------------------------
#define SA_FC 136
__global__ void k_fc1(const float* __restrict__ x, const float* __restrict__ W,
                      const float* __restrict__ bias, const float* __restrict__ g,
                      const float* __restrict__ be, __bf16* __restrict__ h) {
  __shared__ __bf16 sBT[HH * HH];                 // fc1_W^T : [n=128][k=128]
  __shared__ float sBias[HH], sG[HH], sBe[HH];
  __shared__ __bf16 sA[4][16 * SA_FC];
  for (int idx = threadIdx.x; idx < HH * HH; idx += blockDim.x) {
    int k = idx / HH, n = idx % HH;
    sBT[n * HH + k] = (__bf16)W[idx];
  }
  for (int idx = threadIdx.x; idx < HH; idx += blockDim.x) {
    sBias[idx] = bias[idx]; sG[idx] = g[idx]; sBe[idx] = be[idx];
  }
  __syncthreads();
  int lane = threadIdx.x & 31, wave = threadIdx.x >> 5, nw = blockDim.x >> 5;
  __bf16* At = sA[wave];
  for (int tile = blockIdx.x * nw + wave; tile < NN / 16; tile += gridDim.x * nw) {
    int r = lane & 15, hi = lane >> 4;
    const float* xr = x + ((size_t)(tile * 16 + r)) * HH + hi * 64;
    float s = 0.f, s2 = 0.f;
#pragma unroll 8
    for (int t = 0; t < 64; ++t) { float v = xr[t]; s += v; s2 += v * v; }
    s += __shfl_xor(s, 16, 32);
    s2 += __shfl_xor(s2, 16, 32);
    float mu = s * (1.0f / HH);
    float rs = rsqrtf(s2 * (1.0f / HH) - mu * mu + 1e-5f);
#pragma unroll 8
    for (int t = 0; t < 64; ++t) {
      int c = hi * 64 + t;
      At[r * SA_FC + c] = (__bf16)((xr[t] - mu) * rs * sG[c] + sBe[c]);
    }
    LDS_FENCE();
    v8f acc[8] = {};
#pragma unroll
    for (int kc = 0; kc < 4; ++kc) {
      v16bf a = frag_a(At, SA_FC, kc * 32, lane);
#pragma unroll
      for (int nt = 0; nt < 8; ++nt)
        acc[nt] = WMMA_BF16(a, frag_bt(sBT, HH, nt * 16, kc * 32, lane), acc[nt]);
    }
    int n = lane & 15;
#pragma unroll
    for (int rr = 0; rr < 8; ++rr) {
      int row = tile * 16 + rr + 8 * hi;
#pragma unroll
      for (int nt = 0; nt < 8; ++nt) {
        int c = nt * 16 + n;
        h[(size_t)row * HH + c] = (__bf16)gelu_f(acc[nt][rr] + sBias[c]);
      }
    }
    LDS_FENCE();
  }
}

__global__ void k_fc2(const __bf16* __restrict__ h, const float* __restrict__ W,
                      const float* __restrict__ bias, float* __restrict__ x) {
  __shared__ __bf16 sBT[HH * HH];                 // fc2_W^T
  __shared__ float sBias[HH];
  __shared__ __bf16 sA[4][16 * SA_FC];
  for (int idx = threadIdx.x; idx < HH * HH; idx += blockDim.x) {
    int k = idx / HH, n = idx % HH;
    sBT[n * HH + k] = (__bf16)W[idx];
  }
  for (int idx = threadIdx.x; idx < HH; idx += blockDim.x) sBias[idx] = bias[idx];
  __syncthreads();
  int lane = threadIdx.x & 31, wave = threadIdx.x >> 5, nw = blockDim.x >> 5;
  __bf16* At = sA[wave];
  for (int tile = blockIdx.x * nw + wave; tile < NN / 16; tile += gridDim.x * nw) {
    const __bf16* hr = h + (size_t)tile * 16 * HH;
    for (int idx = lane; idx < 16 * HH; idx += 32) {
      int r = idx >> 7, c = idx & 127;
      At[r * SA_FC + c] = hr[r * HH + c];
    }
    LDS_FENCE();
    v8f acc[8] = {};
#pragma unroll
    for (int kc = 0; kc < 4; ++kc) {
      v16bf a = frag_a(At, SA_FC, kc * 32, lane);
#pragma unroll
      for (int nt = 0; nt < 8; ++nt)
        acc[nt] = WMMA_BF16(a, frag_bt(sBT, HH, nt * 16, kc * 32, lane), acc[nt]);
    }
    int n = lane & 15, hi = lane >> 4;
#pragma unroll
    for (int r = 0; r < 8; ++r) {
      int row = tile * 16 + r + 8 * hi;
#pragma unroll
      for (int nt = 0; nt < 8; ++nt) {
        int c = nt * 16 + n;
        x[(size_t)row * HH + c] += acc[nt][r] + sBias[c];
      }
    }
    LDS_FENCE();
  }
}

// ---------------------------------------------------------------------------
extern "C" void kernel_launch(void* const* d_in, const int* in_sizes, int n_in,
                              void* d_out, int out_size, void* d_ws, size_t ws_size,
                              hipStream_t stream) {
  (void)in_sizes; (void)n_in; (void)out_size; (void)ws_size;
  const int*   node_types = (const int*)d_in[0];
  const float* node_feats = (const float*)d_in[1];
  const float* pos        = (const float*)d_in[2];
  const int*   edge_index = (const int*)d_in[3];
  const float* emb_table  = (const float*)d_in[4];
  const float* emb_W      = (const float*)d_in[5];
  const float* emb_b      = (const float*)d_in[6];
  const float* freq       = (const float*)d_in[7];
  const float* gc_ln_g    = (const float*)d_in[8];
  const float* gc_ln_b    = (const float*)d_in[9];
  const float* n2m_W      = (const float*)d_in[10];
  const float* n2m_b      = (const float*)d_in[11];
  const float* r2m_W      = (const float*)d_in[12];
  const float* r2m_b      = (const float*)d_in[13];
  const float* mp1_W      = (const float*)d_in[14];
  const float* mp1_b      = (const float*)d_in[15];
  const float* mp_ln_g    = (const float*)d_in[16];
  const float* mp_ln_b    = (const float*)d_in[17];
  const float* mp2_W      = (const float*)d_in[18];
  const float* mp2_b      = (const float*)d_in[19];
  const float* m2n_W      = (const float*)d_in[20];
  const float* fc_ln_g    = (const float*)d_in[21];
  const float* fc_ln_b    = (const float*)d_in[22];
  const float* fc1_W      = (const float*)d_in[23];
  const float* fc1_b      = (const float*)d_in[24];
  const float* fc2_W      = (const float*)d_in[25];
  const float* fc2_b      = (const float*)d_in[26];

  float* x = (float*)d_out;                                       // [N,128]
  char* ws = (char*)d_ws;
  float*  rbf  = (float*)(ws);                                    // [E,12] f32
  float*  m_in = (float*)(ws + (size_t)EE * RR * 4);              // [N,64] f32
  float*  agg  = (float*)(ws + (size_t)EE * RR * 4 + (size_t)NN * FF * 4);
  __bf16* h    = (__bf16*)(ws + (size_t)EE * RR * 4 + 2ull * NN * FF * 4);  // [N,128] bf16

  k_rbf<<<(EE + 255) / 256, 256, 0, stream>>>(edge_index, pos, freq, rbf);
  k_embed<<<2048, HH, 0, stream>>>(node_types, node_feats, emb_table, emb_W, emb_b, x);
  for (int l = 0; l < LL; ++l) {
    k_n2m<<<1024, 128, 0, stream>>>(x, n2m_W + l * HH * FF, n2m_b + l * FF,
                                    gc_ln_g + l * FF, gc_ln_b + l * FF, m_in);
    k_zero<<<(NN * FF + 255) / 256, 256, 0, stream>>>(agg, NN * FF);
    k_edge<<<4096, 64, 0, stream>>>(edge_index, rbf, m_in,
                                    r2m_W + l * RR * FF, r2m_b + l * FF,
                                    mp1_W + l * 3 * FF * FF, mp1_b + l * FF,
                                    mp_ln_g + l * FF, mp_ln_b + l * FF,
                                    mp2_W + l * FF * FF, mp2_b + l * FF, agg);
    k_m2n<<<1024, 128, 0, stream>>>(agg, m2n_W + l * FF * HH, x);
    k_fc1<<<1024, 128, 0, stream>>>(x, fc1_W + l * HH * HH, fc1_b + l * HH,
                                    fc_ln_g + l * HH, fc_ln_b + l * HH, h);
    k_fc2<<<1024, 128, 0, stream>>>(h, fc2_W + l * HH * HH, fc2_b + l * HH, x);
  }
}
